// CG_36867999269258
// MI455X (gfx1250) — matrix operand order repaired and checked by reading, not compiled, not run on hardware
//
#include <hip/hip_runtime.h>

// ---------------------------------------------------------------------------
// GraphMAE-style GIN encoder pipeline for MI455X (gfx1250, wave32).
//   - GEMMs: v_wmma_f32_16x16x32_bf16, one wave per 16x128 output row-strip
//     (A loaded once; W staged per-block into LDS via global_load_async_to_lds)
//   - Edge aggregation: global_atomic_add_f32 (reference = segment_sum)
//   - BN stats: deterministic two-level column reduction
// ---------------------------------------------------------------------------

#define DD 128
#define NLAYERS 2
#define BN_EPS 1e-5f
#define STAT_BLOCKS 256

typedef __attribute__((ext_vector_type(16))) __bf16        v16bf;
typedef __attribute__((ext_vector_type(8)))  float         v8f;
typedef __attribute__((ext_vector_type(4)))  unsigned int  u32x4;
typedef __attribute__((ext_vector_type(4)))  float         f32x4;

union FragBF { v16bf bf; u32x4 q[2]; };

__device__ __forceinline__ unsigned short f2bf(float f) {
  unsigned int u = __float_as_uint(f);
  unsigned int r = u + 0x7FFFu + ((u >> 16) & 1u);   // round-to-nearest-even
  return (unsigned short)(r >> 16);
}

// ---------------- elementwise helpers ----------------
__global__ void k_copy_f32(const float* __restrict__ x, float* __restrict__ y, int n) {
  int i = blockIdx.x * blockDim.x + threadIdx.x;
  if (i < n) y[i] = x[i];
}

__global__ void k_f32_to_bf16(const float* __restrict__ x, unsigned short* __restrict__ y, int n) {
  int i = blockIdx.x * blockDim.x + threadIdx.x;
  if (i < n) y[i] = f2bf(x[i]);
}

__global__ void k_zero_i32(int* __restrict__ p, int n) {
  int i = blockIdx.x * blockDim.x + threadIdx.x;
  if (i < n) p[i] = 0;
}

// mask/relabel for induced subgraph
__global__ void k_build_mask(const int* __restrict__ mn, int* __restrict__ mask,
                             int* __restrict__ idxmap, int M) {
  int i = blockIdx.x * blockDim.x + threadIdx.x;
  if (i < M) { int n = mn[i]; mask[n] = 1; idxmap[n] = i; }
}

// h[mask_nodes[i], :] = token  (masked-input construction)
__global__ void k_scatter_token(float* __restrict__ h, const int* __restrict__ mn,
                                const float* __restrict__ token, int M) {
  int i = blockIdx.x * blockDim.x + threadIdx.x;
  if (i < M * DD) {
    int node = mn[i >> 7], c = i & 127;
    h[(size_t)node * DD + c] = token[c];
  }
}

// out[i,:] = feat[mask_nodes[i],:]
__global__ void k_gather_rows(const float* __restrict__ feat, const int* __restrict__ mn,
                              float* __restrict__ out, int M) {
  int i = blockIdx.x * blockDim.x + threadIdx.x;
  if (i < M * DD) {
    int node = mn[i >> 7], c = i & 127;
    out[i] = feat[(size_t)node * DD + c];
  }
}

// transpose + convert one 128x128 weight: Wt[n*128+k] = bf16(W[k*128+n])
__global__ void k_conv_wt(const float* __restrict__ W, unsigned short* __restrict__ Wt) {
  int i = blockIdx.x * blockDim.x + threadIdx.x;
  if (i < DD * DD) {
    int k = i >> 7, n = i & 127;
    Wt[n * DD + k] = f2bf(W[i]);
  }
}

// ---------------- edge aggregation (segment_sum) ----------------
// one wave per edge; lane handles 4 contiguous features (coalesced f32x4 read)
__global__ void k_agg_full(const int* __restrict__ src, const int* __restrict__ dst,
                           const float* __restrict__ h, float* __restrict__ out, int E) {
  int wave = (blockIdx.x * blockDim.x + threadIdx.x) >> 5;
  int lane = threadIdx.x & 31;
  if (wave >= E) return;
  int s = src[wave], d = dst[wave];
  f32x4 v = *(const f32x4*)(h + (size_t)s * DD + lane * 4);
  float* op = out + (size_t)d * DD + lane * 4;
  atomicAdd(op + 0, v[0]); atomicAdd(op + 1, v[1]);
  atomicAdd(op + 2, v[2]); atomicAdd(op + 3, v[3]);
}

__global__ void k_agg_sub(const int* __restrict__ src, const int* __restrict__ dst,
                          const int* __restrict__ mask, const int* __restrict__ idxmap,
                          const float* __restrict__ h, float* __restrict__ out, int E) {
  int wave = (blockIdx.x * blockDim.x + threadIdx.x) >> 5;
  int lane = threadIdx.x & 31;
  if (wave >= E) return;
  int s = src[wave], d = dst[wave];
  if (!(mask[s] & mask[d])) return;
  int si = idxmap[s], di = idxmap[d];
  f32x4 v = *(const f32x4*)(h + (size_t)si * DD + lane * 4);
  float* op = out + (size_t)di * DD + lane * 4;
  atomicAdd(op + 0, v[0]); atomicAdd(op + 1, v[1]);
  atomicAdd(op + 2, v[2]); atomicAdd(op + 3, v[3]);
}

// ---------------- WMMA GEMM: Y[R,128] = X[R,128](bf16) * W[128,128] ----------------
// Wt = transposed weight, bf16, Wt[n][k]. Block = 256 threads = 8 waves.
// W (32 KB) is staged once per block into LDS with async global->LDS copies
// (ASYNCcnt path); each wave then computes a full 16x128 row-strip:
// A fragments loaded once from global, B fragments read from LDS, 32 WMMAs/wave.
__global__ void __launch_bounds__(256)
k_gemm_bf16(const unsigned short* __restrict__ X,
            const unsigned short* __restrict__ Wt,
            float* __restrict__ Y, int R) {
  __shared__ unsigned short sW[DD * DD];        // 32 KB bf16 weight tile

  // ---- async stage W into LDS: 256 lanes x 16 B x 8 sweeps = 32 KB ----
  {
    unsigned t = threadIdx.x;
    unsigned ldsaddr = (unsigned)(size_t)(void*)(&sW[t * 8]);          // byte addr in LDS
    unsigned long long gaddr = (unsigned long long)(size_t)Wt + (unsigned long long)t * 16ull;
#pragma unroll
    for (int i = 0; i < 8; ++i) {
      // per-lane 16-byte async copy global -> LDS (ASYNCcnt)
      asm volatile("global_load_async_to_lds_b128 %0, %1, off offset:%2"
                   :: "v"(ldsaddr), "v"(gaddr), "i"(i * 4096) : "memory");
    }
    asm volatile("s_wait_asynccnt 0x0" ::: "memory");
  }
  __syncthreads();

  int wave = (blockIdx.x * blockDim.x + threadIdx.x) >> 5;   // row-strip index tm
  int lane = threadIdx.x & 31;
  int tmRows = (R + 15) >> 4;
  if (wave >= tmRows) return;
  int half = lane >> 4;
  int r    = lane & 15;
  int arow = wave * 16 + r; if (arow > R - 1) arow = R - 1;  // clamp reads
  const unsigned short* xrow = X + (size_t)arow * DD;

  v8f acc[8] = {};
#pragma unroll
  for (int kk = 0; kk < DD; kk += 32) {
    FragBF a;
    // A 16x32 bf16: elem e -> K = kk + half*16 + e (matches B)
    a.q[0] = *(const u32x4*)(xrow + kk + half * 16);
    a.q[1] = *(const u32x4*)(xrow + kk + half * 16 + 8);
#pragma unroll
    for (int tn = 0; tn < 8; ++tn) {
      FragBF b;
      // B 32x16 bf16: elem e -> K = kk + half*16 + e, column n = lane&15 (row of Wt)
      const u32x4* pb = (const u32x4*)(&sW[(tn * 16 + r) * DD + kk + half * 16]);
      b.q[0] = pb[0]; b.q[1] = pb[1];
      acc[tn] = __builtin_amdgcn_wmma_f32_16x16x32_bf16(false, a.bf, false, b.bf,
                                                        (short)0, acc[tn], false, false);
    }
  }

  // ---- epilogue: C/D layout -> row = tm*16 + g + half*8, col = tn*16 + (lane&15) ----
  float* ybase = Y + (size_t)(wave * 16 + half * 8) * DD + r;
  if (wave * 16 + 16 <= R) {                     // full tile (always true for N,M % 16 == 0)
#pragma unroll
    for (int g = 0; g < 8; ++g)
#pragma unroll
      for (int tn = 0; tn < 8; ++tn)
        ybase[(size_t)g * DD + tn * 16] = acc[tn][g];
  } else {                                       // ragged tail (generic-R safety)
#pragma unroll
    for (int g = 0; g < 8; ++g) {
      int row = wave * 16 + g + half * 8;
      if (row < R)
#pragma unroll
        for (int tn = 0; tn < 8; ++tn)
          Y[(size_t)row * DD + tn * 16 + r] = acc[tn][g];
    }
  }
}

// ---------------- BatchNorm (training-mode, biased var) ----------------
// two-level deterministic column reduction
__global__ void k_colstats_partial(const float* __restrict__ x, int R,
                                   float* __restrict__ psum, float* __restrict__ psumsq) {
  int c = threadIdx.x;                           // blockDim.x == 128
  float s = 0.f, ss = 0.f;
  for (int r = blockIdx.x; r < R; r += gridDim.x) {
    float v = x[(size_t)r * DD + c];
    s += v; ss += v * v;
  }
  psum[blockIdx.x * DD + c]   = s;
  psumsq[blockIdx.x * DD + c] = ss;
}

__global__ void k_colstats_final(const float* __restrict__ psum, const float* __restrict__ psumsq,
                                 int R, const float* __restrict__ g, const float* __restrict__ b,
                                 float* __restrict__ scale, float* __restrict__ shift) {
  int c = threadIdx.x;                           // blockDim.x == 128
  float s = 0.f, ss = 0.f;
  for (int i = 0; i < STAT_BLOCKS; ++i) { s += psum[i * DD + c]; ss += psumsq[i * DD + c]; }
  float mean = s / (float)R;
  float var  = ss / (float)R - mean * mean;
  float sc   = g[c] * rsqrtf(var + BN_EPS);
  scale[c] = sc;
  shift[c] = b[c] - mean * sc;
}

__global__ void k_bnrelu(const float* __restrict__ x, const float* __restrict__ scale,
                         const float* __restrict__ shift, float* __restrict__ y, int n) {
  int i = blockIdx.x * blockDim.x + threadIdx.x;
  if (i < n) {
    int c = i & 127;
    y[i] = fmaxf(0.f, x[i] * scale[c] + shift[c]);
  }
}

// ---------------- SCE loss ----------------
// one wave per masked node: cosine between h1[mask_nodes[i]] and h2[i]
__global__ void k_loss_nodes(const float* __restrict__ h1, const int* __restrict__ mn,
                             const float* __restrict__ h2, float* __restrict__ lossbuf, int M) {
  int wave = (blockIdx.x * blockDim.x + threadIdx.x) >> 5;
  int lane = threadIdx.x & 31;
  if (wave >= M) return;
  f32x4 a = *(const f32x4*)(h1 + (size_t)mn[wave] * DD + lane * 4);
  f32x4 b = *(const f32x4*)(h2 + (size_t)wave    * DD + lane * 4);
  float xx = a[0]*a[0] + a[1]*a[1] + a[2]*a[2] + a[3]*a[3];
  float yy = b[0]*b[0] + b[1]*b[1] + b[2]*b[2] + b[3]*b[3];
  float xy = a[0]*b[0] + a[1]*b[1] + a[2]*b[2] + a[3]*b[3];
#pragma unroll
  for (int o = 16; o > 0; o >>= 1) {             // wave32 butterfly reduction
    xx += __shfl_xor(xx, o, 32);
    yy += __shfl_xor(yy, o, 32);
    xy += __shfl_xor(xy, o, 32);
  }
  if (lane == 0) {
    float nx = fmaxf(sqrtf(xx), 1e-12f);
    float ny = fmaxf(sqrtf(yy), 1e-12f);
    lossbuf[wave] = 1.0f - xy / (nx * ny);
  }
}

__global__ void k_reduce_mean(const float* __restrict__ buf, int M, float* __restrict__ out) {
  __shared__ float s[256];
  float acc = 0.f;
  for (int i = threadIdx.x; i < M; i += 256) acc += buf[i];
  s[threadIdx.x] = acc;
  __syncthreads();
  for (int o = 128; o > 0; o >>= 1) {
    if (threadIdx.x < o) s[threadIdx.x] += s[threadIdx.x + o];
    __syncthreads();
  }
  if (threadIdx.x == 0) out[0] = s[0] / (float)M;
}

// ---------------- host orchestration ----------------
static inline int cdiv(long long a, long long b) { return (int)((a + b - 1) / b); }

static void run_encoder(hipStream_t stream, int R, int E,
                        const int* src, const int* dst,
                        const int* mask, const int* idxmap, bool sub,
                        float* hA, float* hB, unsigned short* Xbf,
                        const unsigned short* wt1, const unsigned short* wt2, // + l*DD*DD
                        const float* g1, const float* b1,
                        const float* g2, const float* b2,                     // + l*DD
                        float* psum, float* psumsq, float* scale, float* shift) {
  const int n      = R * DD;
  const int gElem  = cdiv(n, 256);
  const int gEdge  = cdiv((long long)E * 32, 256);
  const int nWaves = (R + 15) >> 4;              // one wave per 16-row strip
  const int gGemm  = cdiv(nWaves, 8);            // 8 waves per 256-thread block

  for (int l = 0; l < NLAYERS; ++l) {
    // agg with self term (GIN eps=0): hB = hA ; hB[dst] += hA[src]
    k_copy_f32<<<gElem, 256, 0, stream>>>(hA, hB, n);
    if (sub)
      k_agg_sub<<<gEdge, 256, 0, stream>>>(src, dst, mask, idxmap, hA, hB, E);
    else
      k_agg_full<<<gEdge, 256, 0, stream>>>(src, dst, hA, hB, E);

    // z1 = (h + agg) @ W1[l]
    k_f32_to_bf16<<<gElem, 256, 0, stream>>>(hB, Xbf, n);
    k_gemm_bf16<<<gGemm, 256, 0, stream>>>(Xbf, wt1 + (size_t)l * DD * DD, hB, R);
    // relu(bn(z1))
    k_colstats_partial<<<STAT_BLOCKS, DD, 0, stream>>>(hB, R, psum, psumsq);
    k_colstats_final<<<1, DD, 0, stream>>>(psum, psumsq, R, g1 + l * DD, b1 + l * DD, scale, shift);
    k_bnrelu<<<gElem, 256, 0, stream>>>(hB, scale, shift, hB, n);
    // z2 = a1 @ W2[l]
    k_f32_to_bf16<<<gElem, 256, 0, stream>>>(hB, Xbf, n);
    k_gemm_bf16<<<gGemm, 256, 0, stream>>>(Xbf, wt2 + (size_t)l * DD * DD, hB, R);
    // h = relu(bn(z2))  -> hA
    k_colstats_partial<<<STAT_BLOCKS, DD, 0, stream>>>(hB, R, psum, psumsq);
    k_colstats_final<<<1, DD, 0, stream>>>(psum, psumsq, R, g2 + l * DD, b2 + l * DD, scale, shift);
    k_bnrelu<<<gElem, 256, 0, stream>>>(hB, scale, shift, hA, n);
  }
}

extern "C" void kernel_launch(void* const* d_in, const int* in_sizes, int n_in,
                              void* d_out, int out_size, void* d_ws, size_t ws_size,
                              hipStream_t stream) {
  const int N = in_sizes[0] / DD;
  const int E = in_sizes[2] / 2;
  const int M = in_sizes[3];

  const float* feat    = (const float*)d_in[0];
  const float* token   = (const float*)d_in[1];
  const int*   src     = (const int*)d_in[2];
  const int*   dst     = src + E;
  const int*   mn      = (const int*)d_in[3];
  const float* on_W1   = (const float*)d_in[4];
  const float* on_W2   = (const float*)d_in[5];
  const float* on_g1   = (const float*)d_in[6];
  const float* on_b1   = (const float*)d_in[7];
  const float* on_g2   = (const float*)d_in[8];
  const float* on_b2   = (const float*)d_in[9];
  const float* tg_W1   = (const float*)d_in[10];
  const float* tg_W2   = (const float*)d_in[11];
  const float* tg_g1   = (const float*)d_in[12];
  const float* tg_b1   = (const float*)d_in[13];
  const float* tg_g2   = (const float*)d_in[14];
  const float* tg_b2   = (const float*)d_in[15];
  float* out = (float*)d_out;

  // --- workspace carve-up (256-B aligned) ---
  char* base = (char*)d_ws;
  size_t off = 0;
  auto carve = [&](size_t bytes) -> char* {
    char* p = base + off;
    off = (off + bytes + 255) & ~(size_t)255;
    return p;
  };
  float*          hA     = (float*)carve((size_t)N * DD * 4); // online activations (persist)
  float*          hB     = (float*)carve((size_t)N * DD * 4); // online scratch
  unsigned short* Xbf    = (unsigned short*)carve((size_t)N * DD * 2);
  float*          hAt    = (float*)carve((size_t)M * DD * 4); // target activations
  float*          hBt    = (float*)carve((size_t)M * DD * 4);
  unsigned short* wt     = (unsigned short*)carve((size_t)8 * DD * DD * 2); // bf16 W^T, 8 mats
  float*          psum   = (float*)carve((size_t)STAT_BLOCKS * DD * 4);
  float*          psumsq = (float*)carve((size_t)STAT_BLOCKS * DD * 4);
  float*          scale  = (float*)carve(DD * 4);
  float*          shift  = (float*)carve(DD * 4);
  int*            mask   = (int*)carve((size_t)N * 4);
  int*            idxmap = (int*)carve((size_t)N * 4);
  float*          lossb  = (float*)carve((size_t)M * 4);
  (void)ws_size; (void)n_in; (void)out_size;

  // --- setup: subgraph relabel, masked input, target input, bf16 weights ---
  k_zero_i32<<<cdiv(N, 256), 256, 0, stream>>>(mask, N);
  k_build_mask<<<cdiv(M, 256), 256, 0, stream>>>(mn, mask, idxmap, M);

  k_copy_f32<<<cdiv((long long)N * DD, 256), 256, 0, stream>>>(feat, hA, N * DD);
  k_scatter_token<<<cdiv((long long)M * DD, 256), 256, 0, stream>>>(hA, mn, token, M);
  k_gather_rows<<<cdiv((long long)M * DD, 256), 256, 0, stream>>>(feat, mn, hAt, M);

  const int gW = cdiv(DD * DD, 256);
  const float* wsrc[8] = { on_W1, on_W1 + DD * DD, on_W2, on_W2 + DD * DD,
                           tg_W1, tg_W1 + DD * DD, tg_W2, tg_W2 + DD * DD };
  for (int i = 0; i < 8; ++i)
    k_conv_wt<<<gW, 256, 0, stream>>>(wsrc[i], wt + (size_t)i * DD * DD);

  // --- online encoder (full graph, masked inputs) ---
  run_encoder(stream, N, E, src, dst, nullptr, nullptr, false,
              hA, hB, Xbf,
              wt + 0 * DD * DD, wt + 2 * DD * DD,
              on_g1, on_b1, on_g2, on_b2,
              psum, psumsq, scale, shift);

  // --- target encoder (induced subgraph on mask nodes) ---
  run_encoder(stream, M, E, src, dst, mask, idxmap, true,
              hAt, hBt, Xbf,
              wt + 4 * DD * DD, wt + 6 * DD * DD,
              tg_g1, tg_b1, tg_g2, tg_b2,
              psum, psumsq, scale, shift);

  // --- SCE loss: mean(1 - cos(h1[mask], h2)) ---
  k_loss_nodes<<<cdiv((long long)M * 32, 256), 256, 0, stream>>>(hA, mn, hAt, lossb, M);
  k_reduce_mean<<<1, 256, 0, stream>>>(lossb, M, out);
}
